// LlamaAttention_heavy_hitter_16552803958787
// MI455X (gfx1250) — matrix operand order repaired and checked by reading, not compiled, not run on hardware
//
#include <hip/hip_runtime.h>
#include <math.h>

// ---------------- problem constants ----------------
constexpr int SEQ     = 2048;
constexpr int HIDN    = 4096;
constexpr int NHEAD   = 32;
constexpr int DHEAD   = 128;
constexpr int HBGT    = 204;   // int(0.1 * 2048)
constexpr int RECENTW = 204;   // int(0.1 * 2048)
constexpr float NEGV  = -3.402823466e38f;   // jnp.finfo(f32).min
constexpr float SCALE = 0.08838834764831845f; // 1/sqrt(128)

typedef __bf16 bf16;
typedef __attribute__((ext_vector_type(16))) __bf16 v16bf;
typedef __attribute__((ext_vector_type(8)))  float  v8f;

// ---------------- WMMA fragment loaders (gfx1250 wave32 layouts) ----------------
// A-matrix 16x32 bf16 (M x K), row-major source, K contiguous.
// lane L (<16): row M=L, elements e=0..15 hold K = e (e<8) / e+8 (e>=8)
// lane L (>=16): row M=L-16, K shifted by +8. => two 16B chunks at +8g and +16+8g.
static __device__ __forceinline__ v16bf load_a_frag(const bf16* A, int ldA,
                                                    int mBase, int kBase, int lane) {
  int g = lane >> 4;
  const bf16* p = A + (size_t)(mBase + (lane & 15)) * ldA + kBase + 8 * g;
  union { uint4 u[2]; v16bf v; } t;
  t.u[0] = *reinterpret_cast<const uint4*>(p);
  t.u[1] = *reinterpret_cast<const uint4*>(p + 16);
  return t.v;
}

// B-matrix 32x16 bf16 (K x N) from a K-major ("transposed") source Bt[k][n].
// lane L holds column N = L&15; elements e hold K = e+8g (e<8) / e+8+8g (e>=8), g = L>>4.
static __device__ __forceinline__ v16bf load_b_frag(const bf16* Bt, int ldB,
                                                    int kBase, int nBase, int lane) {
  int g = lane >> 4;
  const bf16* p = Bt + (size_t)(kBase + 8 * g) * ldB + nBase + (lane & 15);
  v16bf v;
  for (int s = 0; s < 8; ++s) {
    v[s] = p[(size_t)s * ldB];
    v[8 + s] = p[(size_t)(16 + s) * ldB];
  }
  return v;
}

static __device__ __forceinline__ v8f wmma_bf16(v16bf a, v16bf b, v8f c) {
  return __builtin_amdgcn_wmma_f32_16x16x32_bf16(false, a, false, b, (short)0, c,
                                                 false, false);
}

// ---------------- 1) f32 -> bf16 convert ----------------
__global__ void k_cvt_bf16(const float* __restrict__ in, bf16* __restrict__ out, int n) {
  int i = blockIdx.x * blockDim.x + threadIdx.x;
  if (i < n) out[i] = (bf16)in[i];
}

// ---------------- 2) weight transpose + convert: Wt[k][n] = (bf16)W[n][k] --------
__global__ void k_wtrans(const float* __restrict__ W, bf16* __restrict__ Wt) {
  __shared__ float tile[32][33];
  int nt = blockIdx.x % (HIDN / 32);
  int kt = blockIdx.x / (HIDN / 32);
  int n0 = nt * 32, k0 = kt * 32;
  int tx = threadIdx.x, ty = threadIdx.y;
  for (int i = 0; i < 4; ++i)
    tile[ty + 8 * i][tx] = W[(size_t)(n0 + ty + 8 * i) * HIDN + k0 + tx];
  __syncthreads();
  for (int i = 0; i < 4; ++i)
    Wt[(size_t)(k0 + ty + 8 * i) * HIDN + n0 + tx] = (bf16)tile[tx][ty + 8 * i];
}

// ---------------- 3) generic NT GEMM: C[m][n] = sum_k A[m][k] * Bt[k][n] ---------
// one wave -> 32(M) x 64(N) tile. Two-stage software pipeline with *named*
// stage variables (no runtime-indexed vector arrays -> no scratch spills).
// Requires K % 64 == 0 (true here: K == 4096). blockDim 128 (4 waves).
__global__ void __launch_bounds__(128)
k_gemm_nt(const bf16* __restrict__ A, int ldA,
          const bf16* __restrict__ Bt, int ldB,
          float* __restrict__ C, int ldC, int N64, int K) {
  int lane = threadIdx.x & 31, wid = threadIdx.x >> 5;
  int idx = blockIdx.x * 4 + wid;
  int mT = idx / N64, nQ = idx % N64;
  int mBase = mT * 32, nBase = nQ * 64;
  v8f zero = {};
  v8f acc0[4] = {zero, zero, zero, zero};
  v8f acc1[4] = {zero, zero, zero, zero};

  // stage A preload (kb = 0)
  v16bf a0A = load_a_frag(A, ldA, mBase,      0, lane);
  v16bf a1A = load_a_frag(A, ldA, mBase + 16, 0, lane);
  v16bf b0A = load_b_frag(Bt, ldB, 0, nBase,      lane);
  v16bf b1A = load_b_frag(Bt, ldB, 0, nBase + 16, lane);
  v16bf b2A = load_b_frag(Bt, ldB, 0, nBase + 32, lane);
  v16bf b3A = load_b_frag(Bt, ldB, 0, nBase + 48, lane);

  for (int kb = 0; kb < K; kb += 64) {
    // issue stage-B loads (kb+32 < K because K % 64 == 0)
    v16bf a0B = load_a_frag(A, ldA, mBase,      kb + 32, lane);
    v16bf a1B = load_a_frag(A, ldA, mBase + 16, kb + 32, lane);
    v16bf b0B = load_b_frag(Bt, ldB, kb + 32, nBase,      lane);
    v16bf b1B = load_b_frag(Bt, ldB, kb + 32, nBase + 16, lane);
    v16bf b2B = load_b_frag(Bt, ldB, kb + 32, nBase + 32, lane);
    v16bf b3B = load_b_frag(Bt, ldB, kb + 32, nBase + 48, lane);

    // consume stage A
    acc0[0] = wmma_bf16(a0A, b0A, acc0[0]);
    acc1[0] = wmma_bf16(a1A, b0A, acc1[0]);
    acc0[1] = wmma_bf16(a0A, b1A, acc0[1]);
    acc1[1] = wmma_bf16(a1A, b1A, acc1[1]);
    acc0[2] = wmma_bf16(a0A, b2A, acc0[2]);
    acc1[2] = wmma_bf16(a1A, b2A, acc1[2]);
    acc0[3] = wmma_bf16(a0A, b3A, acc0[3]);
    acc1[3] = wmma_bf16(a1A, b3A, acc1[3]);

    // refill stage A (kb+64) while stage B is in flight
    if (kb + 64 < K) {
      a0A = load_a_frag(A, ldA, mBase,      kb + 64, lane);
      a1A = load_a_frag(A, ldA, mBase + 16, kb + 64, lane);
      b0A = load_b_frag(Bt, ldB, kb + 64, nBase,      lane);
      b1A = load_b_frag(Bt, ldB, kb + 64, nBase + 16, lane);
      b2A = load_b_frag(Bt, ldB, kb + 64, nBase + 32, lane);
      b3A = load_b_frag(Bt, ldB, kb + 64, nBase + 48, lane);
    }

    // consume stage B
    acc0[0] = wmma_bf16(a0B, b0B, acc0[0]);
    acc1[0] = wmma_bf16(a1B, b0B, acc1[0]);
    acc0[1] = wmma_bf16(a0B, b1B, acc0[1]);
    acc1[1] = wmma_bf16(a1B, b1B, acc1[1]);
    acc0[2] = wmma_bf16(a0B, b2B, acc0[2]);
    acc1[2] = wmma_bf16(a1B, b2B, acc1[2]);
    acc0[3] = wmma_bf16(a0B, b3B, acc0[3]);
    acc1[3] = wmma_bf16(a1B, b3B, acc1[3]);
  }

  for (int i = 0; i < 4; ++i) {
    int n = nBase + 16 * i + (lane & 15);
    for (int r = 0; r < 8; ++r) {
      int m0 = mBase + r + 8 * (lane >> 4);
      C[(size_t)m0 * ldC + n] = acc0[i][r];
      C[(size_t)(m0 + 16) * ldC + n] = acc1[i][r];
    }
  }
}

// ---------------- 4) RoPE + repack to per-head bf16 layouts ----------------------
// QB[h][s][d], KT[h][d][s] (transposed for scores-B), VB[h][s][d]
__global__ void k_rope_repack(const float* __restrict__ QF, const float* __restrict__ KF,
                              const float* __restrict__ VF,
                              bf16* __restrict__ QB, bf16* __restrict__ KT,
                              bf16* __restrict__ VB) {
  size_t idx = (size_t)blockIdx.x * blockDim.x + threadIdx.x;
  int s = (int)(idx / HIDN), hd = (int)(idx % HIDN);
  int h = hd / DHEAD, d = hd % DHEAD;
  float q = QF[idx], k = KF[idx], v = VF[idx];
  float rq = (d < 64) ? -QF[idx + 64] : QF[idx - 64];
  float rk = (d < 64) ? -KF[idx + 64] : KF[idx - 64];
  // inv_freq = 10000^(-2*(d%64)/128)
  float inv = __expf(-(2.0f * (float)(d & 63) / 128.0f) * 9.210340371976184f);
  float ang = (float)s * inv;
  float sn, cs;
  __sincosf(ang, &sn, &cs);
  float qr = q * cs + rq * sn;
  float kr = k * cs + rk * sn;
  QB[((size_t)h * SEQ + s) * DHEAD + d] = (bf16)qr;
  KT[((size_t)h * DHEAD + d) * SEQ + s] = (bf16)kr;
  VB[((size_t)h * SEQ + s) * DHEAD + d] = (bf16)v;
}

// ---------------- 5) scores = QK^T * scale + causal NEG --------------------------
__global__ void __launch_bounds__(128)
k_scores(const bf16* __restrict__ QB, const bf16* __restrict__ KT,
         float* __restrict__ SC) {
  int lane = threadIdx.x & 31, wid = threadIdx.x >> 5;
  int idx = blockIdx.x * 4 + wid;
  const int quadsPerHead = (SEQ / 16) * (SEQ / 64);
  int h = idx / quadsPerHead, rem = idx % quadsPerHead;
  int mT = rem / (SEQ / 64), nQ = rem % (SEQ / 64);
  int mBase = mT * 16, nBase = nQ * 64;
  float* out = SC + (size_t)h * SEQ * SEQ;
  if (nBase > mBase + 15) {  // strip fully above diagonal
    for (int i = 0; i < 4; ++i) {
      int n = nBase + 16 * i + (lane & 15);
      for (int r = 0; r < 8; ++r) {
        int m = mBase + r + 8 * (lane >> 4);
        out[(size_t)m * SEQ + n] = NEGV;
      }
    }
    return;
  }
  const bf16* A  = QB + (size_t)h * SEQ * DHEAD;
  const bf16* Bt = KT + (size_t)h * DHEAD * SEQ;
  v8f zero = {};
  v8f acc[4] = {zero, zero, zero, zero};
  for (int kb = 0; kb < DHEAD; kb += 32) {
    v16bf a = load_a_frag(A, DHEAD, mBase, kb, lane);
    for (int i = 0; i < 4; ++i) {
      v16bf b = load_b_frag(Bt, SEQ, kb, nBase + 16 * i, lane);
      acc[i] = wmma_bf16(a, b, acc[i]);
    }
  }
  for (int i = 0; i < 4; ++i) {
    int n = nBase + 16 * i + (lane & 15);
    for (int r = 0; r < 8; ++r) {
      int m = mBase + r + 8 * (lane >> 4);
      out[(size_t)m * SEQ + n] = (n <= m) ? acc[i][r] * SCALE : NEGV;
    }
  }
}

// ---------------- 6) sequential H2O heavy-hitter scan (one WG per head) ----------
__global__ void __launch_bounds__(1024)
k_h2o_scan(const float* __restrict__ SC, unsigned* __restrict__ MSK) {
  int h = blockIdx.x;
  int t = threadIdx.x, lane = t & 31, wid = t >> 5;
  __shared__ float accL[SEQ];
  __shared__ float red[32];
  __shared__ int ired[32];
  __shared__ int wpfx[32];
  __shared__ unsigned mw[SEQ / 32];
  __shared__ float fmaxs, fsum;
  __shared__ int icnt, cgt;
  const float* base = SC + (size_t)h * SEQ * SEQ;
  unsigned* mbase = MSK + (size_t)h * SEQ * (SEQ / 32);
  accL[t] = 0.f; accL[t + 1024] = 0.f;
  __syncthreads();
  int j0 = 2 * t, j1 = j0 + 1;

  // phase 1: acc0 = sum of softmax of first HBGT rows, cols >= HBGT zeroed
  for (int r = 0; r < HBGT; ++r) {
    const float* row = base + (size_t)r * SEQ;
    __builtin_prefetch(row + SEQ + j0, 0, 1);  // next row
    float x0 = row[j0], x1 = row[j1];
    float m = fmaxf(x0, x1);
    for (int o = 16; o; o >>= 1) m = fmaxf(m, __shfl_xor(m, o));
    if (!lane) red[wid] = m;
    __syncthreads();
    if (!t) { float mm = red[0]; for (int i = 1; i < 32; ++i) mm = fmaxf(mm, red[i]); fmaxs = mm; }
    __syncthreads();
    float e0 = __expf(x0 - fmaxs), e1 = __expf(x1 - fmaxs);
    float s = e0 + e1;
    for (int o = 16; o; o >>= 1) s += __shfl_xor(s, o);
    if (!lane) red[wid] = s;
    __syncthreads();
    if (!t) { float ss = 0.f; for (int i = 0; i < 32; ++i) ss += red[i]; fsum = ss; }
    __syncthreads();
    float inv = 1.f / fsum;
    if (j0 < HBGT) accL[j0] += e0 * inv;
    if (j1 < HBGT) accL[j1] += e1 * inv;
  }
  // heavy mask for first HBGT rows: col < HBGT
  for (int i = t; i < HBGT * (SEQ / 32); i += 1024) {
    int w = i & (SEQ / 32 - 1);
    unsigned val = (w < HBGT / 32) ? 0xFFFFFFFFu
                 : ((w == HBGT / 32) ? ((1u << (HBGT & 31)) - 1u) : 0u);
    mbase[i] = val;
  }

  // phase 2: sequential rows HBGT..SEQ-1
  for (int r = HBGT; r < SEQ; ++r) {
    const float* row = base + (size_t)r * SEQ;
    __builtin_prefetch(row + SEQ + j0, 0, 1);  // next row while we reduce this one
    float x0 = row[j0], x1 = row[j1];
    float m = fmaxf(x0, x1);
    for (int o = 16; o; o >>= 1) m = fmaxf(m, __shfl_xor(m, o));
    if (!lane) red[wid] = m;
    __syncthreads();
    if (!t) { float mm = red[0]; for (int i = 1; i < 32; ++i) mm = fmaxf(mm, red[i]); fmaxs = mm; }
    __syncthreads();
    float e0 = __expf(x0 - fmaxs), e1 = __expf(x1 - fmaxs);
    float s = e0 + e1;
    for (int o = 16; o; o >>= 1) s += __shfl_xor(s, o);
    if (!lane) red[wid] = s;
    __syncthreads();
    if (!t) { float ss = 0.f; for (int i = 0; i < 32; ++i) ss += red[i]; fsum = ss; }
    __syncthreads();
    float inv = 1.f / fsum;
    float p0 = e0 * inv, p1 = e1 * inv;

    // radix-select the (HBGT-1)-th largest of acc (acc >= 0 -> uint bits monotone)
    float a0 = accL[j0], a1 = accL[j1];
    unsigned u0 = __float_as_uint(a0), u1 = __float_as_uint(a1);
    unsigned T = 0u;
    for (int bit = 30; bit >= 0; --bit) {
      unsigned cand = T | (1u << bit);
      int c = (u0 >= cand) + (u1 >= cand);
      for (int o = 16; o; o >>= 1) c += __shfl_xor(c, o);
      if (!lane) ired[wid] = c;
      __syncthreads();
      if (!t) { int cc = 0; for (int i = 0; i < 32; ++i) cc += ired[i]; icnt = cc; }
      __syncthreads();
      if (icnt >= HBGT - 1) T = cand;
    }
    int g0 = (u0 > T), g1 = (u1 > T);
    int c = g0 + g1;
    for (int o = 16; o; o >>= 1) c += __shfl_xor(c, o);
    if (!lane) ired[wid] = c;
    __syncthreads();
    if (!t) { int cc = 0; for (int i = 0; i < 32; ++i) cc += ired[i]; cgt = cc; }
    __syncthreads();
    int need = (HBGT - 1) - cgt;

    // ties (acc == T) taken in ascending index order (top_k tie order)
    int e0i = (u0 == T), e1i = (u1 == T);
    unsigned b0 = (unsigned)__ballot(e0i);
    unsigned b1 = (unsigned)__ballot(e1i);
    int wcnt = __popc(b0) + __popc(b1);
    if (!lane) ired[wid] = wcnt;
    __syncthreads();
    if (!t) { int run = 0; for (int i = 0; i < 32; ++i) { wpfx[i] = run; run += ired[i]; } }
    __syncthreads();
    unsigned ltm = (1u << lane) - 1u;
    int rank0 = wpfx[wid] + __popc(b0 & ltm) + __popc(b1 & ltm);
    int rank1 = rank0 + e0i;
    int sel0 = g0 || (e0i && rank0 < need) || (j0 == r);
    int sel1 = g1 || (e1i && rank1 < need) || (j1 == r);
    accL[j0] = (a0 + p0) * (sel0 ? 1.f : 0.f);
    accL[j1] = (a1 + p1) * (sel1 ? 1.f : 0.f);

    if (t < SEQ / 32) mw[t] = 0u;
    __syncthreads();
    if (sel0) atomicOr(&mw[j0 >> 5], 1u << (j0 & 31));
    if (sel1) atomicOr(&mw[j1 >> 5], 1u << (j1 & 31));
    __syncthreads();
    if (t < SEQ / 32) mbase[(size_t)r * (SEQ / 32) + t] = mw[t];
    __syncthreads();
  }
}

// ------- 7) final mask + softmax, pack probs bf16 in-place over f32 score rows ---
__global__ void __launch_bounds__(1024)
k_mask_softmax_pack(float* __restrict__ SC, const unsigned* __restrict__ MSK) {
  int row = blockIdx.x;
  int r = row % SEQ;
  float* rowp = SC + (size_t)row * SEQ;
  int t = threadIdx.x, lane = t & 31, wid = t >> 5;
  __shared__ float red[32];
  __shared__ float bmax, bsum;
  int j0 = 2 * t, j1 = j0 + 1;
  float x0 = rowp[j0], x1 = rowp[j1];
  unsigned w = MSK[(size_t)row * (SEQ / 32) + (j0 >> 5)];
  bool k0 = (j0 <= r) && (((w >> (j0 & 31)) & 1u) || (j0 >= r - RECENTW));
  bool k1 = (j1 <= r) && (((w >> (j1 & 31)) & 1u) || (j1 >= r - RECENTW));
  x0 = k0 ? x0 : NEGV;
  x1 = k1 ? x1 : NEGV;
  float m = fmaxf(x0, x1);
  for (int o = 16; o; o >>= 1) m = fmaxf(m, __shfl_xor(m, o));
  if (!lane) red[wid] = m;
  __syncthreads();
  if (!t) { float mm = red[0]; for (int i = 1; i < 32; ++i) mm = fmaxf(mm, red[i]); bmax = mm; }
  __syncthreads();
  float e0 = __expf(x0 - bmax), e1 = __expf(x1 - bmax);
  float s = e0 + e1;
  for (int o = 16; o; o >>= 1) s += __shfl_xor(s, o);
  if (!lane) red[wid] = s;
  __syncthreads();
  if (!t) { float ss = 0.f; for (int i = 0; i < 32; ++i) ss += red[i]; bsum = ss; }
  __syncthreads();
  float inv = 1.f / bsum;
  union { bf16 b[2]; unsigned u; } pk;
  pk.b[0] = (bf16)(e0 * inv);
  pk.b[1] = (bf16)(e1 * inv);
  reinterpret_cast<unsigned*>(rowp)[t] = pk.u;  // all reads done before reduction barriers
}

// ---------------- 8) P @ V (bf16 WMMA), write AO[s][h*D+d] -----------------------
__global__ void __launch_bounds__(128)
k_pv(const float* __restrict__ SC, const bf16* __restrict__ VB, bf16* __restrict__ AO) {
  int lane = threadIdx.x & 31, wid = threadIdx.x >> 5;
  int idx = blockIdx.x * 4 + wid;
  const int tilesPerHead = (SEQ / 16) * (DHEAD / 16);
  int h = idx / tilesPerHead, rem = idx % tilesPerHead;
  int mT = rem / (DHEAD / 16), dT = rem % (DHEAD / 16);
  int mBase = mT * 16, dBase = dT * 16;
  const bf16* A  = reinterpret_cast<const bf16*>(SC + (size_t)h * SEQ * SEQ); // packed rows, stride 2*SEQ bf16
  const bf16* Bv = VB + (size_t)h * SEQ * DHEAD;
  int kEnd = (mBase + 16 + 31) & ~31;
  if (kEnd > SEQ) kEnd = SEQ;
  v8f acc = {};
  for (int kb = 0; kb < kEnd; kb += 32) {
    v16bf a = load_a_frag(A, 2 * SEQ, mBase, kb, lane);
    v16bf b = load_b_frag(Bv, DHEAD, kb, dBase, lane);
    acc = wmma_bf16(a, b, acc);
  }
  int n = dBase + (lane & 15);
  for (int r = 0; r < 8; ++r) {
    int m = mBase + r + 8 * (lane >> 4);
    AO[(size_t)m * HIDN + h * DHEAD + n] = (bf16)acc[r];
  }
}

// ---------------- host launch ----------------------------------------------------
extern "C" void kernel_launch(void* const* d_in, const int* in_sizes, int n_in,
                              void* d_out, int out_size, void* d_ws, size_t ws_size,
                              hipStream_t stream) {
  (void)in_sizes; (void)n_in; (void)out_size; (void)ws_size;
  const float* hs = (const float*)d_in[0];
  // d_in[1] = attention_mask (causal, recomputed), d_in[2] = position_ids (= arange)
  const float* wq = (const float*)d_in[3];
  const float* wk = (const float*)d_in[4];
  const float* wv = (const float*)d_in[5];
  const float* wo = (const float*)d_in[6];
  float* out = (float*)d_out;

  char* w = (char*)d_ws;
  auto carve = [&](size_t bytes) {
    char* p = w;
    w += (bytes + 255) & ~(size_t)255;
    return p;
  };
  bf16* HSB = (bf16*)carve((size_t)SEQ * HIDN * 2);
  bf16* WQT = (bf16*)carve((size_t)HIDN * HIDN * 2);
  bf16* WKT = (bf16*)carve((size_t)HIDN * HIDN * 2);
  bf16* WVT = (bf16*)carve((size_t)HIDN * HIDN * 2);
  bf16* WOT = (bf16*)carve((size_t)HIDN * HIDN * 2);
  float* QF = (float*)carve((size_t)SEQ * HIDN * 4);
  float* KF = (float*)carve((size_t)SEQ * HIDN * 4);
  float* VF = (float*)carve((size_t)SEQ * HIDN * 4);
  bf16* QB = (bf16*)carve((size_t)NHEAD * SEQ * DHEAD * 2);
  bf16* KT = (bf16*)carve((size_t)NHEAD * DHEAD * SEQ * 2);
  bf16* VB = (bf16*)carve((size_t)NHEAD * SEQ * DHEAD * 2);
  float* SCp = (float*)carve((size_t)NHEAD * SEQ * SEQ * 4);
  unsigned* MSKp = (unsigned*)carve((size_t)NHEAD * SEQ * (SEQ / 32) * 4);
  bf16* AO = (bf16*)carve((size_t)SEQ * HIDN * 2);

  // 1) convert hidden states
  k_cvt_bf16<<<(SEQ * HIDN) / 256, 256, 0, stream>>>(hs, HSB, SEQ * HIDN);
  // 2) transpose + convert weights
  dim3 tb(32, 8);
  int tgrid = (HIDN / 32) * (HIDN / 32);
  k_wtrans<<<tgrid, tb, 0, stream>>>(wq, WQT);
  k_wtrans<<<tgrid, tb, 0, stream>>>(wk, WKT);
  k_wtrans<<<tgrid, tb, 0, stream>>>(wv, WVT);
  k_wtrans<<<tgrid, tb, 0, stream>>>(wo, WOT);
  // 3) Q/K/V projections (32x64 per wave, 4 waves per block)
  int gblocks = ((SEQ / 32) * (HIDN / 64)) / 4;
  k_gemm_nt<<<gblocks, 128, 0, stream>>>(HSB, HIDN, WQT, HIDN, QF, HIDN, HIDN / 64, HIDN);
  k_gemm_nt<<<gblocks, 128, 0, stream>>>(HSB, HIDN, WKT, HIDN, KF, HIDN, HIDN / 64, HIDN);
  k_gemm_nt<<<gblocks, 128, 0, stream>>>(HSB, HIDN, WVT, HIDN, VF, HIDN, HIDN / 64, HIDN);
  // 4) RoPE + repack
  k_rope_repack<<<(SEQ * HIDN) / 256, 256, 0, stream>>>(QF, KF, VF, QB, KT, VB);
  // 5) scores
  int sblocks = (NHEAD * (SEQ / 16) * (SEQ / 64)) / 4;
  k_scores<<<sblocks, 128, 0, stream>>>(QB, KT, SCp);
  // 6) sequential H2O scan (one workgroup per head)
  k_h2o_scan<<<NHEAD, 1024, 0, stream>>>(SCp, MSKp);
  // 7) final mask + softmax, pack bf16 probs in place
  k_mask_softmax_pack<<<NHEAD * SEQ, 1024, 0, stream>>>(SCp, MSKp);
  // 8) P @ V
  int pvblocks = (NHEAD * (SEQ / 16) * (DHEAD / 16)) / 4;
  k_pv<<<pvblocks, 128, 0, stream>>>(SCp, VB, AO);
  // 9) output projection -> d_out
  k_gemm_nt<<<gblocks, 128, 0, stream>>>(AO, HIDN, WOT, HIDN, out, HIDN, HIDN / 64, HIDN);
}
